// DCNv2_2233382994219
// MI455X (gfx1250) — compile-verified, matrix operand order checked by reading, not run on hardware
//
#include <hip/hip_runtime.h>
#include <hip/hip_bf16.h>

typedef __attribute__((ext_vector_type(16))) _Float16 v16h;
typedef __attribute__((ext_vector_type(2)))  _Float16 h2;
typedef __attribute__((ext_vector_type(8)))  float    v8f;

#define Hs 128
#define Ws 128
#define CIN 64
#define COUT 64
#define KTAPS 9
#define NWFRAG (KTAPS * 2 * 4 * 32 * 16)   // 36864 f16 = 72 KB

// ---------------------------------------------------------------------------
// Prologue: reorder weight (COUT, CIN, 3, 3) fp32 -> WMMA-B-fragment-contiguous
// f16 in workspace. Layout: wr[(k*8 + c*4 + t)*32 + lane][e], e = 0..15,
// where for V_WMMA_F32_16X16X32_F16 B-fragment: n = t*16 + (lane&15),
// kk (reduction row within 32-chunk) = (lane>>4)*16 + e, cin = c*32 + kk.
// ---------------------------------------------------------------------------
__global__ void dcnv2_reorder_w(const float* __restrict__ w,
                                _Float16* __restrict__ wr) {
    int i = blockIdx.x * 256 + threadIdx.x;
    if (i >= NWFRAG) return;
    int e    = i & 15;
    int lane = (i >> 4) & 31;
    int t    = (i >> 9) & 3;
    int c    = (i >> 11) & 1;
    int k    = (i >> 12);            // 0..8
    int cout = t * 16 + (lane & 15);
    int cin  = c * 32 + (lane >> 4) * 16 + e;
    wr[i] = (_Float16)w[(cout * CIN + cin) * KTAPS + k];
}

// ---------------------------------------------------------------------------
// Main fused kernel: one block per (b, ho) output row. 256 threads = 8 waves.
// Block prologue: cooperative one-shot copy of all 72 KB of reordered weights
//                 into LDS (amortized over 576 WMMAs per block).
// Phase 1 per tap: bilinear gather -> LDS f16 tile [128 pix][64 cin].
// Phase 2 per tap: hoist 2 A-fragments + 8 B-fragments into registers
//                  (clustered ds_load_b128), then 8 back-to-back WMMAs.
// ---------------------------------------------------------------------------
__global__ __launch_bounds__(256) void dcnv2_main(
    const float* __restrict__ x,      // (B, CIN, H, W)
    const float* __restrict__ off,    // (B, 2*K, H, W)
    const float* __restrict__ msk,    // (B, K, H, W)
    const _Float16* __restrict__ wr,  // reordered weights (see above)
    const float* __restrict__ bias,   // (COUT)
    float* __restrict__ out)          // (B, COUT, H, W)
{
    __shared__ alignas(32) _Float16 wlds[NWFRAG];        // 72 KB weights
    __shared__ alignas(32) _Float16 vals[2][Ws * CIN];   // 2 x 16 KB A tiles

    const int blk  = blockIdx.x;        // b*128 + ho
    const int b    = blk >> 7;
    const int ho   = blk & 127;
    const int tid  = threadIdx.x;
    const int lane = tid & 31;
    const int wave = tid >> 5;

    // One-shot cooperative weight preload into LDS (ordered by the first
    // per-tap __syncthreads before any phase-2 read).
    {
        const uint4* src = (const uint4*)wr;
        uint4*       dst = (uint4*)wlds;
        const int nvec = NWFRAG * 2 / 16;   // 4608 x 16B
#pragma unroll
        for (int i = 0; i < nvec / 256; ++i)
            dst[i * 256 + tid] = src[i * 256 + tid];
    }

    // phase-1 role: pixel p1 (=wo), channel half
    const int p1   = tid & 127;
    const int half = tid >> 7;          // 0 or 1 -> cin 0..31 / 32..63
    // phase-2 role: wave M-tile base (wo base)
    const int m0   = wave * 16;

    // accumulators, init with bias (each lane's 8 elems share the same cout)
    v8f acc[4];
    const int ncol = lane & 15;
#pragma unroll
    for (int t = 0; t < 4; ++t) {
        float bv = bias[t * 16 + ncol];
#pragma unroll
        for (int i = 0; i < 8; ++i) acc[t][i] = bv;
    }

    const float* xb = x   + (size_t)b * CIN * Hs * Ws;
    const float* ob = off + (size_t)b * 2 * KTAPS * Hs * Ws;
    const float* mb = msk + (size_t)b * KTAPS * Hs * Ws;

    // per-lane LDS addresses (invariant across taps)
    const int aoff = (m0 + (lane & 15)) * CIN + (lane >> 4) * 16;

    for (int k = 0; k < KTAPS; ++k) {
        const int buf = k & 1;

        // ---------------- Phase 1: gather tap k into LDS ----------------
        {
            const int kh = k / 3, kw = k % 3;
            const int rowoff = ho * Ws + p1;
            const float py = (float)(ho + kh - 1) + ob[(2 * k + 0) * Hs * Ws + rowoff];
            const float px = (float)(p1 + kw - 1) + ob[(2 * k + 1) * Hs * Ws + rowoff];
            const float mv = mb[k * Hs * Ws + rowoff];

            const float fy = floorf(py), fx = floorf(px);
            const int   y0 = (int)fy,    x0 = (int)fx;
            const int   y1 = y0 + 1,     x1 = x0 + 1;
            const float ly = py - fy,    lx = px - fx;
            const float hy = 1.0f - ly,  hx = 1.0f - lx;

            const bool iy0 = (y0 >= 0) & (y0 < Hs);
            const bool iy1 = (y1 >= 0) & (y1 < Hs);
            const bool ix0 = (x0 >= 0) & (x0 < Ws);
            const bool ix1 = (x1 >= 0) & (x1 < Ws);
            const float w00 = (iy0 & ix0) ? hy * hx * mv : 0.0f;
            const float w01 = (iy0 & ix1) ? hy * lx * mv : 0.0f;
            const float w10 = (iy1 & ix0) ? ly * hx * mv : 0.0f;
            const float w11 = (iy1 & ix1) ? ly * lx * mv : 0.0f;

            const int y0c = min(max(y0, 0), Hs - 1);
            const int y1c = min(max(y1, 0), Hs - 1);
            const int x0c = min(max(x0, 0), Ws - 1);
            const int x1c = min(max(x1, 0), Ws - 1);
            const int i00 = y0c * Ws + x0c;
            const int i01 = y0c * Ws + x1c;
            const int i10 = y1c * Ws + x0c;
            const int i11 = y1c * Ws + x1c;

            const float* xc = xb + (size_t)(half * 32) * Hs * Ws;
            h2* dst2 = (h2*)&vals[buf][p1 * CIN + half * 32];
#pragma unroll 4
            for (int c = 0; c < 32; c += 2) {
                const float* xp0 = xc + (size_t)c * Hs * Ws;
                const float* xp1 = xp0 + Hs * Ws;
                float v0 = xp0[i00] * w00 + xp0[i01] * w01 +
                           xp0[i10] * w10 + xp0[i11] * w11;
                float v1 = xp1[i00] * w00 + xp1[i01] * w01 +
                           xp1[i10] * w10 + xp1[i11] * w11;
                h2 pk = { (_Float16)v0, (_Float16)v1 };
                dst2[c >> 1] = pk;
            }
        }

        __syncthreads();   // LDS tile (and, on k==0, weight preload) ready

        // ---------------- Phase 2: hoist fragments, then 8 WMMAs ----------
        // A-fragments (16x32 f16): lane l -> row m0+(l&15), 16 consecutive
        // cin starting at c*32 + (l>>4)*16  (contiguous k order, same as B).
        const v16h a0 = *(const v16h*)&vals[buf][aoff];
        const v16h a1 = *(const v16h*)&vals[buf][aoff + 32];

        v16h bf[8];
#pragma unroll
        for (int i = 0; i < 8; ++i)        // i = c*4 + t
            bf[i] = *(const v16h*)&wlds[(size_t)((k * 8 + i) * 32 + lane) * 16];

#pragma unroll
        for (int t = 0; t < 4; ++t)
            acc[t] = __builtin_amdgcn_wmma_f32_16x16x32_f16(
                false, a0, false, bf[t], (short)0, acc[t], false, false);
#pragma unroll
        for (int t = 0; t < 4; ++t)
            acc[t] = __builtin_amdgcn_wmma_f32_16x16x32_f16(
                false, a1, false, bf[4 + t], (short)0, acc[t], false, false);
    }

    // ---------------- Epilogue: write C tiles ------------------------------
    // C layout: VGPR v, lanes 0-15 -> (m=v, n=lane), lanes 16-31 -> (m=8+v).
    // m maps to wo (contiguous) -> one 32B vector store per tile per lane.
    const int mrow = (lane >> 4) * 8;
#pragma unroll
    for (int t = 0; t < 4; ++t) {
        const int n = t * 16 + ncol;
        float* o = out + ((size_t)(b * COUT + n) * Hs + ho) * Ws + m0 + mrow;
        *(v8f*)o = acc[t];
    }
}

extern "C" void kernel_launch(void* const* d_in, const int* in_sizes, int n_in,
                              void* d_out, int out_size, void* d_ws, size_t ws_size,
                              hipStream_t stream) {
    const float* x      = (const float*)d_in[0];
    const float* offset = (const float*)d_in[1];
    const float* mask   = (const float*)d_in[2];
    const float* weight = (const float*)d_in[3];
    const float* bias   = (const float*)d_in[4];
    float* out = (float*)d_out;
    _Float16* wr = (_Float16*)d_ws;   // 36864 f16 = 72 KB of workspace

    dcnv2_reorder_w<<<(NWFRAG + 255) / 256, 256, 0, stream>>>(weight, wr);

    const int B = 8;
    dcnv2_main<<<B * Hs, 256, 0, stream>>>(x, offset, mask, wr, bias, out);
}
